// ann1_coding_snn7_59992103190942
// MI455X (gfx1250) — matrix operand order repaired and checked by reading, not compiled, not run on hardware
//
#include <hip/hip_runtime.h>

// ---------------------------------------------------------------------------
// CDNA5 WMMA (wave32): v_wmma_f32_16x16x32_bf16
// ---------------------------------------------------------------------------
typedef __attribute__((ext_vector_type(16))) __bf16          v16bf;
typedef __attribute__((ext_vector_type(16))) unsigned short  v16us;
typedef __attribute__((ext_vector_type(8)))  float           v8f;

#define TLEN  20
#define NB    64
#define SIGMA 0.77880078f   // exp(-1/tau_m), tau_m = 4
#define B1C   0.86972940f   // v0 * (exp(-1/4) - exp(-1))

__device__ __forceinline__ unsigned short f2bf(float x) {
  unsigned u = __float_as_uint(x);
  unsigned r = (u + 0x7FFFu + ((u >> 16) & 1u)) >> 16;   // round-to-nearest-even
  return (unsigned short)r;
}
__device__ __forceinline__ float bf2f(unsigned short u) {
  return __uint_as_float(((unsigned)u) << 16);
}

// deterministic N(0,1): splitmix64 + Box-Muller (stand-in for jax threefry)
__device__ __forceinline__ float gauss1(unsigned long long key) {
  unsigned long long z = key * 0x9E3779B97F4A7C15ull + 0xD1B54A32D192ED03ull;
  z = (z ^ (z >> 30)) * 0xBF58476D1CE4E5B9ull;
  z = (z ^ (z >> 27)) * 0x94D049BB133111EBull;
  z ^= (z >> 31);
  unsigned u1 = (unsigned)z, u2 = (unsigned)(z >> 32);
  float f1 = ((float)u1 + 0.5f) * (1.0f / 4294967296.0f);
  float f2 = ((float)u2) * (1.0f / 4294967296.0f);
  float r = __fsqrt_rn(-2.0f * __logf(f1));
  return r * __cosf(6.28318530718f * f2);
}

// ---------------------------------------------------------------------------
// Prep kernels (run once per call, tiny)
// ---------------------------------------------------------------------------
// f32 [OCn][K] -> bf16 [OCp][Kp], zero padded
__global__ __launch_bounds__(256) void prep_w_kernel(
    const float* __restrict__ src, unsigned short* __restrict__ dst,
    int OCn, int K, int OCp, int Kp) {
  int i = blockIdx.x * blockDim.x + threadIdx.x;
  if (i >= OCp * Kp) return;
  int oc = i / Kp, k = i - oc * Kp;
  float v = (oc < OCn && k < K) ? src[(size_t)oc * K + k] : 0.0f;
  dst[i] = f2bf(v);
}

// f32 -> bf16 elementwise (network input image)
__global__ __launch_bounds__(256) void prep_x_kernel(
    const float* __restrict__ src, unsigned short* __restrict__ dst, int n) {
  int i = blockIdx.x * blockDim.x + threadIdx.x;
  if (i < n) dst[i] = f2bf(src[i]);
}

// im2col BYTE-offset table for a 3x3 VALID conv (bf16 elements, x2):
//   offs[k] = 2*(ic*IH*IW + r*IW + s)
// padded entries (k >= K) -> 0 (their weights are zero, product is zero)
__global__ __launch_bounds__(256) void prep_offs_kernel(
    unsigned* __restrict__ dst, int K, int Kp, int IW, int IHIW) {
  int k = blockIdx.x * blockDim.x + threadIdx.x;
  if (k >= Kp) return;
  unsigned o = 0;
  if (k < K) {
    int ic = k / 9;
    int rs = k - ic * 9;
    int r  = rs / 3;
    int s  = rs - r * 3;
    o = (unsigned)(ic * IHIW + r * IW + s) * 2u;
  }
  dst[k] = o;
}

// ---------------------------------------------------------------------------
// Implicit-GEMM conv / FC via v_wmma_f32_16x16x32_bf16.
//   C[p][oc] = sum_k im2col(X)[p][k] * W[oc][k] + bias[oc]
// One wave computes a 16(M) x (16*NT)(N) strip; 8 waves per block.
// A fragment (lane L, elem i): m = L&15, k = kb + (i&7) + (i>>3)*16 + (L>=16)*8
//   -> two k-contiguous runs of 8: byte offsets via uint4 table loads (KS=3)
//      or two aligned b128 activation loads (KS=1). Activations are bf16.
// B fragment (lane L, elem i): n = L&15, k = kb + (i&7) + (i>>3)*16 + (L>=16)*8 -> 2 x b128
// C fragment (lane L, vgpr j): m = j + (L>=16)*8, n = L&15
// All addressing: uniform 64-bit base + unsigned 32-bit offset (GVS mode).
// ---------------------------------------------------------------------------
template <int KS, int NT>
__global__ __launch_bounds__(256) void wmma_gemm_kernel(
    const unsigned short* __restrict__ Xb, const unsigned short* __restrict__ Wb,
    const unsigned* __restrict__ offs, const float* __restrict__ bias,
    float* __restrict__ Y,
    int IC, int IHIW, int IW, int OH, int OW, int OCs, int Kp,
    int oc_n, int Mtiles) {
  const int wave = threadIdx.x >> 5;
  const int lane = threadIdx.x & 31;
  const int hi   = lane >> 4;
  const int lm   = lane & 15;
  const int mtile = blockIdx.x * 8 + wave;
  if (mtile >= Mtiles) return;               // wave-uniform: EXEC stays all-1s
  const int OHW = OH * OW;

  // geometry of the A-row this lane gathers (lanes L and L+16 share a row)
  const int m   = mtile * 16 + lm;
  const int img = m / OHW;
  const int rem = m - img * OHW;
  const int oh  = rem / OW;
  const int ow  = rem - oh * OW;

  const char* Xc = (const char*)Xb;
  const char* Wc = (const char*)Wb;
  // per-lane 32-bit byte offsets off uniform bases
  const unsigned rowb  = ((unsigned)img * (unsigned)(IC * IHIW) +
                          (unsigned)(KS == 1 ? 0 : (oh * IW + ow))) * 2u;
  const unsigned nrow[NT > 0 ? NT : 1] = {};  // (placeholder, computed below)
  (void)nrow;

  v8f acc[NT];
#pragma unroll
  for (int u = 0; u < NT; ++u)
    acc[u] = (v8f){0.f, 0.f, 0.f, 0.f, 0.f, 0.f, 0.f, 0.f};

  unsigned wbase[NT];
#pragma unroll
  for (int u = 0; u < NT; ++u) {
    unsigned n = (unsigned)((blockIdx.y * NT + u) * 16 + lm);
    wbase[u] = (n * (unsigned)Kp + (unsigned)(hi << 3)) * 2u;
  }

  for (int kb = 0; kb < Kp; kb += 32) {
    // ---- A fragment
    v16us au;
    if constexpr (KS == 1) {
      union { uint4 q[2]; v16us v; } ax;
      unsigned ab = rowb + (unsigned)((kb + (hi << 3)) * 2);
      ax.q[0] = *(const uint4*)(Xc + ab);
      ax.q[1] = *(const uint4*)(Xc + ab + 32u);
      au = ax.v;
    } else {
      const int k0 = kb + (hi << 3);
      uint4 o0 = *(const uint4*)(offs + k0);
      uint4 o1 = *(const uint4*)(offs + k0 + 4);
      uint4 o2 = *(const uint4*)(offs + k0 + 16);
      uint4 o3 = *(const uint4*)(offs + k0 + 20);
      au[0]  = *(const unsigned short*)(Xc + (rowb + o0.x));
      au[1]  = *(const unsigned short*)(Xc + (rowb + o0.y));
      au[2]  = *(const unsigned short*)(Xc + (rowb + o0.z));
      au[3]  = *(const unsigned short*)(Xc + (rowb + o0.w));
      au[4]  = *(const unsigned short*)(Xc + (rowb + o1.x));
      au[5]  = *(const unsigned short*)(Xc + (rowb + o1.y));
      au[6]  = *(const unsigned short*)(Xc + (rowb + o1.z));
      au[7]  = *(const unsigned short*)(Xc + (rowb + o1.w));
      au[8]  = *(const unsigned short*)(Xc + (rowb + o2.x));
      au[9]  = *(const unsigned short*)(Xc + (rowb + o2.y));
      au[10] = *(const unsigned short*)(Xc + (rowb + o2.z));
      au[11] = *(const unsigned short*)(Xc + (rowb + o2.w));
      au[12] = *(const unsigned short*)(Xc + (rowb + o3.x));
      au[13] = *(const unsigned short*)(Xc + (rowb + o3.y));
      au[14] = *(const unsigned short*)(Xc + (rowb + o3.z));
      au[15] = *(const unsigned short*)(Xc + (rowb + o3.w));
    }
    v16bf a = __builtin_bit_cast(v16bf, au);

    // ---- NT B fragments + WMMAs
#pragma unroll
    for (int u = 0; u < NT; ++u) {
      union { uint4 q[2]; v16us v; } bw;
      unsigned wb_off = wbase[u] + (unsigned)(kb * 2);
      bw.q[0] = *(const uint4*)(Wc + wb_off);
      bw.q[1] = *(const uint4*)(Wc + wb_off + 32u);
      v16bf b = __builtin_bit_cast(v16bf, bw.v);
      acc[u] = __builtin_amdgcn_wmma_f32_16x16x32_bf16(
          false, a, false, b, (short)0, acc[u], false, false);
    }
  }

  // ---- store: scatter C back to NCHW (f32 pre-activations)
  {
    int mm0  = mtile * 16 + (hi << 3);
    int img0 = mm0 / OHW;
    int rm0  = mm0 - img0 * OHW;
#pragma unroll
    for (int u = 0; u < NT; ++u) {
      const int n = (blockIdx.y * NT + u) * 16 + lm;
      if (n < oc_n) {                        // snn8: only 10 of 16 cols real
        float bv = bias[n];
        int img2 = img0, rm2 = rm0;
#pragma unroll
        for (int j = 0; j < 8; ++j) {
          unsigned yi = ((unsigned)img2 * (unsigned)OCs + (unsigned)n) *
                            (unsigned)OHW + (unsigned)rm2;
          Y[yi] = acc[u][j] + bv;
          rm2++;
          int wrap = (rm2 == OHW);
          img2 += wrap;
          rm2 = wrap ? 0 : rm2;
        }
      }
    }
  }
}

// ---------------------------------------------------------------------------
// VAE coding + dual_exp_iir(ax2): per (b,c,h,w), serial over t in registers.
// ann (f32) holds conv output [b][64][30][30]; mu=relu(ch<32),
// std=exp(0.5*relu(ch+32)). Writes bf16 psp2 as [(t*NB+b)][32][30][30].
// ---------------------------------------------------------------------------
__global__ __launch_bounds__(256) void coding_iir_kernel(
    const float* __restrict__ ann, const float* __restrict__ a1a,
    const float* __restrict__ a2a, unsigned short* __restrict__ psp) {
  int i = blockIdx.x * blockDim.x + threadIdx.x;
  const int F = 32 * 900;
  if (i >= NB * F) return;
  unsigned b  = (unsigned)i / F;
  unsigned f  = (unsigned)i - b * F;
  unsigned c  = f / 900;
  unsigned hw = f - c * 900;
  float mu = fmaxf(ann[(b * 64u + c) * 900u + hw], 0.0f);
  float lv = fmaxf(ann[(b * 64u + 32u + c) * 900u + hw], 0.0f);
  float sd = __expf(0.5f * lv);
  float a1 = a1a[f], a2 = a2a[f];
  float p1 = 0.f, p2 = 0.f;
  unsigned idx = (b * 32u + c) * 900u + hw;
  const unsigned stride = (unsigned)NB * 32u * 900u;
  for (int t = 0; t < TLEN; ++t) {
    unsigned long long key = ((unsigned long long)t * NB + b) * (unsigned long long)F + f;
    float xt = mu + gauss1(key) * sd;
    float y  = a1 * p1 + a2 * p2 + B1C * xt;
    p2 = p1; p1 = y;
    psp[idx] = f2bf(y);
    idx += stride;
  }
}

// ---------------------------------------------------------------------------
// LIF + dual_exp_iir fused: f32 pre-activations [(t*NB+b)][F] -> bf16 psp.
// Each thread owns one neuron, loops t in registers.
// ---------------------------------------------------------------------------
__global__ __launch_bounds__(256) void lif_iir_kernel(
    const float* __restrict__ in, unsigned short* __restrict__ outp,
    const float* __restrict__ a1a, const float* __restrict__ a2a, int F) {
  int i = blockIdx.x * blockDim.x + threadIdx.x;
  if (i >= NB * F) return;
  unsigned f = (unsigned)i % (unsigned)F;
  float a1 = a1a[f], a2 = a2a[f];
  float v = 0.f, s = 0.f, p1 = 0.f, p2 = 0.f;
  unsigned idx = (unsigned)i;
  const unsigned stride = (unsigned)NB * (unsigned)F;
  for (int t = 0; t < TLEN; ++t) {
    float pt = in[idx];
    v = SIGMA * v * (1.0f - s) + pt;
    s = (v > 1.0f) ? 1.0f : 0.0f;
    float y = a1 * p1 + a2 * p2 + B1C * s;
    p2 = p1; p1 = y;
    outp[idx] = f2bf(y);
    idx += stride;
  }
}

// ---------------------------------------------------------------------------
// 2x2 maxpool (VALID) + dual_exp_iir fused, bf16 in / bf16 out.
// ---------------------------------------------------------------------------
__global__ __launch_bounds__(256) void pool_iir_kernel(
    const unsigned short* __restrict__ in, unsigned short* __restrict__ out,
    const float* __restrict__ a1a, const float* __restrict__ a2a,
    int C, int H, int W, int H2, int W2) {
  int i = blockIdx.x * blockDim.x + threadIdx.x;
  const int F = C * H2 * W2;
  if (i >= NB * F) return;
  unsigned b  = (unsigned)i / (unsigned)F;
  unsigned f  = (unsigned)i - b * (unsigned)F;
  unsigned c  = f / (unsigned)(H2 * W2);
  unsigned r  = f - c * (unsigned)(H2 * W2);
  unsigned oh = r / (unsigned)W2;
  unsigned ow = r - oh * (unsigned)W2;
  float a1 = a1a[f], a2 = a2a[f], p1 = 0.f, p2 = 0.f;
  unsigned ibase = ((b * C + c) * H + 2u * oh) * W + 2u * ow;
  unsigned obase = (b * C + c) * (unsigned)(H2 * W2) + r;
  const unsigned istr = (unsigned)NB * (unsigned)(C * H * W);
  const unsigned ostr = (unsigned)NB * (unsigned)F;
  for (int t = 0; t < TLEN; ++t) {
    float m = fmaxf(fmaxf(bf2f(in[ibase]), bf2f(in[ibase + 1])),
                    fmaxf(bf2f(in[ibase + W]), bf2f(in[ibase + W + 1])));
    float y = a1 * p1 + a2 * p2 + B1C * m;
    p2 = p1; p1 = y;
    out[obase] = f2bf(y);
    ibase += istr;
    obase += ostr;
  }
}

// ---------------------------------------------------------------------------
// Final LIF over snn8 pre-activations (f32) [(t*NB+b)][10] -> out[b][10][t]
// ---------------------------------------------------------------------------
__global__ __launch_bounds__(256) void final_lif_kernel(
    const float* __restrict__ x, float* __restrict__ out) {
  int i = blockIdx.x * blockDim.x + threadIdx.x;
  if (i >= NB * 10) return;
  unsigned b = (unsigned)i / 10u, o = (unsigned)i - b * 10u;
  float v = 0.f, s = 0.f;
  unsigned idx = b * 10u + o;
  for (int t = 0; t < TLEN; ++t) {
    float pt = x[idx];
    v = SIGMA * v * (1.0f - s) + pt;
    s = (v > 1.0f) ? 1.0f : 0.0f;
    out[(b * 10u + o) * TLEN + (unsigned)t] = s;
    idx += (unsigned)NB * 10u;
  }
}

// ---------------------------------------------------------------------------
// Host-side orchestration
// ---------------------------------------------------------------------------
extern "C" void kernel_launch(void* const* d_in, const int* in_sizes, int n_in,
                              void* d_out, int out_size, void* d_ws, size_t ws_size,
                              hipStream_t stream) {
  const float* inputs  = (const float*)d_in[0];
  const float* ann1_w  = (const float*)d_in[1];
  const float* ann1_b  = (const float*)d_in[2];
  const float* snn2_w  = (const float*)d_in[3];
  const float* snn2_b  = (const float*)d_in[4];
  const float* snn3_w  = (const float*)d_in[5];
  const float* snn3_b  = (const float*)d_in[6];
  const float* snn5_w  = (const float*)d_in[7];
  const float* snn5_b  = (const float*)d_in[8];
  const float* snn7_w  = (const float*)d_in[9];
  const float* snn7_b  = (const float*)d_in[10];
  const float* snn8_w  = (const float*)d_in[11];
  const float* snn8_b  = (const float*)d_in[12];
  const float* ax2_a1  = (const float*)d_in[13];
  const float* ax2_a2  = (const float*)d_in[14];
  const float* ax3_a1  = (const float*)d_in[15];
  const float* ax3_a2  = (const float*)d_in[16];
  const float* ax4_a1  = (const float*)d_in[17];
  const float* ax4_a2  = (const float*)d_in[18];
  const float* ax5_a1  = (const float*)d_in[19];
  const float* ax5_a2  = (const float*)d_in[20];
  const float* ax6_a1  = (const float*)d_in[21];
  const float* ax6_a2  = (const float*)d_in[22];
  const float* ax7_a1  = (const float*)d_in[23];
  const float* ax7_a2  = (const float*)d_in[24];
  const float* ax8_a1  = (const float*)d_in[25];
  const float* ax8_a2  = (const float*)d_in[26];
  float* out = (float*)d_out;

  // ---- workspace carve-up ----------------------------------------------
  // CF: f32 pre-activation scratch (max conv out = 1280*64*26*26)
  float* CF = (float*)d_ws;                       // 55,377,920 f32
  unsigned* offs_ann = (unsigned*)(CF + 55377920);//   32
  unsigned* offs_2   = offs_ann + 32;             //  288
  unsigned* offs_3   = offs_2 + 288;              //  288
  unsigned* offs_5   = offs_3 + 288;              //  576  (total 1184 = 4736 B)
  unsigned short* P0 = (unsigned short*)(offs_5 + 576);  // 55,377,920 bf16
  unsigned short* P1 = P0 + 55377920;                    // 32,112,640 bf16
  unsigned short* wb_ann = P1 + 32112640;
  unsigned short* wb2 = wb_ann + 64 * 32;
  unsigned short* wb3 = wb2 + 32 * 288;
  unsigned short* wb5 = wb3 + 64 * 288;
  unsigned short* wb7 = wb5 + 64 * 576;
  unsigned short* wb8 = wb7 + 512 * 1600;
  unsigned short* xin = wb8 + 16 * 512;           // 196,608 bf16 (input image)

  // ---- 1. one-time conversions -----------------------------------------
  auto prep = [&](const float* s, unsigned short* d, int OCn, int K, int OCp, int Kp) {
    int n = OCp * Kp;
    prep_w_kernel<<<(n + 255) / 256, 256, 0, stream>>>(s, d, OCn, K, OCp, Kp);
  };
  prep(ann1_w, wb_ann, 64, 27, 64, 32);
  prep(snn2_w, wb2, 32, 288, 32, 288);
  prep(snn3_w, wb3, 64, 288, 64, 288);
  prep(snn5_w, wb5, 64, 576, 64, 576);
  prep(snn7_w, wb7, 512, 1600, 512, 1600);
  prep(snn8_w, wb8, 10, 512, 16, 512);
  prep_x_kernel<<<(196608 + 255) / 256, 256, 0, stream>>>(inputs, xin, 196608);
  prep_offs_kernel<<<1, 256, 0, stream>>>(offs_ann, 27, 32, 32, 1024);
  prep_offs_kernel<<<2, 256, 0, stream>>>(offs_2, 288, 288, 30, 900);
  prep_offs_kernel<<<2, 256, 0, stream>>>(offs_3, 288, 288, 28, 784);
  prep_offs_kernel<<<3, 256, 0, stream>>>(offs_5, 576, 576, 13, 169);

  // ---- 2. ann1 conv (3->64 on 32x32) : WMMA -> CF ----------------------
  {
    int Mt = (64 * 900) / 16;                            // 3600
    wmma_gemm_kernel<3, 4><<<dim3(Mt / 8, 1), 256, 0, stream>>>(
        xin, wb_ann, offs_ann, ann1_b, CF, 3, 1024, 32, 30, 30, 64, 32, 64, Mt);
  }
  // ---- 3. coding + IIR ax2 -> psp2 (bf16) in P0 ------------------------
  {
    int n = NB * 32 * 900;
    coding_iir_kernel<<<(n + 255) / 256, 256, 0, stream>>>(CF, ax2_a1, ax2_a2, P0);
  }
  // ---- 4. conv snn2 (32->32 on 30x30) : WMMA -> CF ---------------------
  {
    int Mt = (TLEN * NB * 28 * 28) / 16;                 // 62720
    wmma_gemm_kernel<3, 2><<<dim3(Mt / 8, 1), 256, 0, stream>>>(
        P0, wb2, offs_2, snn2_b, CF, 32, 900, 30, 28, 28, 32, 288, 32, Mt);
  }
  // ---- 5. LIF + IIR ax3 -> psp3 (bf16) in P1 ---------------------------
  {
    int F = 32 * 28 * 28, n = NB * F;
    lif_iir_kernel<<<(n + 255) / 256, 256, 0, stream>>>(CF, P1, ax3_a1, ax3_a2, F);
  }
  // ---- 6. conv snn3 (32->64 on 28x28) : WMMA -> CF ---------------------
  {
    int Mt = (TLEN * NB * 26 * 26) / 16;                 // 54080
    wmma_gemm_kernel<3, 4><<<dim3(Mt / 8, 1), 256, 0, stream>>>(
        P1, wb3, offs_3, snn3_b, CF, 32, 784, 28, 26, 26, 64, 288, 64, Mt);
  }
  // ---- 7. LIF + IIR ax4 -> psp4 (bf16) in P0 ---------------------------
  {
    int F = 64 * 26 * 26, n = NB * F;
    lif_iir_kernel<<<(n + 255) / 256, 256, 0, stream>>>(CF, P0, ax4_a1, ax4_a2, F);
  }
  // ---- 8. maxpool 26->13 + IIR ax5 -> psp5 (bf16) in P1 ----------------
  {
    int n = NB * 64 * 13 * 13;
    pool_iir_kernel<<<(n + 255) / 256, 256, 0, stream>>>(P0, P1, ax5_a1, ax5_a2,
        64, 26, 26, 13, 13);
  }
  // ---- 9. conv snn5 (64->64 on 13x13) : WMMA -> CF ---------------------
  {
    int Mt = (TLEN * NB * 11 * 11) / 16;                 // 9680
    wmma_gemm_kernel<3, 4><<<dim3(Mt / 8, 1), 256, 0, stream>>>(
        P1, wb5, offs_5, snn5_b, CF, 64, 169, 13, 11, 11, 64, 576, 64, Mt);
  }
  // ---- 10. LIF + IIR ax6 -> psp6 (bf16) in P0 --------------------------
  {
    int F = 64 * 11 * 11, n = NB * F;
    lif_iir_kernel<<<(n + 255) / 256, 256, 0, stream>>>(CF, P0, ax6_a1, ax6_a2, F);
  }
  // ---- 11. maxpool 11->5 + IIR ax7 -> psp7 (bf16, [p][1600]) in P1 -----
  {
    int n = NB * 64 * 5 * 5;
    pool_iir_kernel<<<(n + 255) / 256, 256, 0, stream>>>(P0, P1, ax7_a1, ax7_a2,
        64, 11, 11, 5, 5);
  }
  // ---- 12. FC snn7 (1600->512) : WMMA -> CF ----------------------------
  {
    int Mt = (TLEN * NB) / 16;                           // 80
    wmma_gemm_kernel<1, 4><<<dim3(Mt / 8, 8), 256, 0, stream>>>(
        P1, wb7, (const unsigned*)nullptr, snn7_b, CF, 1600, 1, 1, 1, 1, 512, 1600, 512, Mt);
  }
  // ---- 13. LIF + IIR ax8 -> psp8 (bf16) in P0 --------------------------
  {
    int F = 512, n = NB * F;
    lif_iir_kernel<<<(n + 255) / 256, 256, 0, stream>>>(CF, P0, ax8_a1, ax8_a2, F);
  }
  // ---- 14. FC snn8 (512->10, N padded to 16) : WMMA -> CF --------------
  {
    int Mt = (TLEN * NB) / 16;                           // 80
    wmma_gemm_kernel<1, 1><<<dim3(Mt / 8, 1), 256, 0, stream>>>(
        P0, wb8, (const unsigned*)nullptr, snn8_b, CF, 512, 1, 1, 1, 1, 10, 512, 10, Mt);
  }
  // ---- 15. final LIF + transpose to (B,10,T) ---------------------------
  {
    int n = NB * 10;
    final_lif_kernel<<<(n + 255) / 256, 256, 0, stream>>>(CF, out);
  }
}
